// MtMs_21500606284435
// MI455X (gfx1250) — matrix operand order repaired and checked by reading, not compiled, not run on hardware
//
#include <hip/hip_runtime.h>

#define DIN   512
#define DH    1024
#define DOUT  512
#define TNUM  8
#define PDIM  64
#define NROW  1024

#define PAD_ROWS 1152   // 72 tiles * 16 (worst case padding: 1024 + 8*15 <= 1144)
#define NTILES   72
#define PLANE    (DIN * DH)   // 524288 elements, same for both weight planes

typedef __attribute__((ext_vector_type(16))) __bf16 v16bf;
typedef __attribute__((ext_vector_type(8)))  __bf16 v8bf;
typedef __attribute__((ext_vector_type(8)))  float  v8f;

// ---------------------------------------------------------------------------
// 1) Counting sort of rows by task (single block). Produces:
//    perm[PAD_ROWS]   : padded-slot -> source row (-1 for padding)
//    tile_task[NTILES]: 16-row tile -> task id (-1 past the end)
// ---------------------------------------------------------------------------
__global__ void mtms_sort_kernel(const int* __restrict__ tasks,
                                 int* __restrict__ perm,
                                 int* __restrict__ tile_task) {
    __shared__ int cnt[TNUM];
    __shared__ int off[TNUM + 1];
    __shared__ int cur[TNUM];
    int tid = threadIdx.x;
    if (tid < TNUM) { cnt[tid] = 0; cur[tid] = 0; }
    __syncthreads();
    for (int i = tid; i < NROW; i += blockDim.x)
        atomicAdd(&cnt[tasks[i]], 1);
    __syncthreads();
    if (tid == 0) {
        int o = 0;
        for (int t = 0; t < TNUM; ++t) { off[t] = o; o += (cnt[t] + 15) & ~15; }
        off[TNUM] = o;
    }
    __syncthreads();
    for (int i = tid; i < PAD_ROWS; i += blockDim.x) perm[i] = -1;
    for (int i = tid; i < NTILES; i += blockDim.x) {
        int base = i * 16, t = -1;
        for (int tt = 0; tt < TNUM; ++tt)
            if (base >= off[tt] && base < off[tt + 1]) t = tt;
        tile_task[i] = t;
    }
    __syncthreads();
    for (int i = tid; i < NROW; i += blockDim.x) {
        int t = tasks[i];
        int pos = atomicAdd(&cur[t], 1);
        perm[off[t] + pos] = i;
    }
}

// ---------------------------------------------------------------------------
// 2) Gather x rows into task-grouped padded layout, f32 -> bf16
// ---------------------------------------------------------------------------
__global__ void mtms_gather_kernel(const float* __restrict__ x,
                                   const int* __restrict__ perm,
                                   __bf16* __restrict__ Xg) {
    int idx = blockIdx.x * blockDim.x + threadIdx.x;   // PAD_ROWS*DIN/4 threads
    int row = idx >> 7;            // / (DIN/4)
    int col = (idx & 127) << 2;
    int src = perm[row];
    float4 v = make_float4(0.f, 0.f, 0.f, 0.f);
    if (src >= 0) v = *(const float4*)(x + (size_t)src * DIN + col);
    union { __bf16 b[4]; unsigned long long q; } o;
    o.b[0] = (__bf16)v.x; o.b[1] = (__bf16)v.y; o.b[2] = (__bf16)v.z; o.b[3] = (__bf16)v.w;
    *(unsigned long long*)(Xg + (size_t)row * DIN + col) = o.q;
}

// ---------------------------------------------------------------------------
// 3) Bias hypernetwork (tiny) + mesa A-fragment precompute.
//    bias{1,2}[t][j] = b[j] + mb_b[j] + mb[j,:].e[t,:]
//    eAfrag[pbTile][lane][16]: bf16 WMMA A fragment of mesa_emb (M=t, K=p),
//    identical for every hypernet wave -> computed once here.
// ---------------------------------------------------------------------------
__global__ void mtms_bias_kernel(const float* __restrict__ b1, const float* __restrict__ b2,
                                 const float* __restrict__ mb1, const float* __restrict__ mb1_b,
                                 const float* __restrict__ mb2, const float* __restrict__ mb2_b,
                                 const float* __restrict__ emb,
                                 float* __restrict__ bias1, float* __restrict__ bias2,
                                 __bf16* __restrict__ eAfrag) {
    int j = blockIdx.x * blockDim.x + threadIdx.x;
    if (j < 64) {        // 2 pb-tiles * 32 lanes
        int lane = j & 31;
        int t = lane & 15, hiA = (lane >> 4) * 8, pb = (j >> 5) * 32;
        for (int e = 0; e < 16; ++e) {
            int p = pb + e + 2 * hiA;   // contiguous 16-bit A K-layout (k = 16h + e)
            float f = (t < TNUM) ? emb[t * PDIM + p] : 0.f;
            eAfrag[j * 16 + e] = (__bf16)f;
        }
    }
    if (j < DH) {
        float base = b1[j] + mb1_b[j];
        for (int t = 0; t < TNUM; ++t) {
            float acc = base;
            for (int p = 0; p < PDIM; ++p) acc += mb1[(size_t)j * PDIM + p] * emb[t * PDIM + p];
            bias1[t * DH + j] = acc;
        }
    } else if (j < DH + DOUT) {
        int jj = j - DH;
        float base = b2[jj] + mb2_b[jj];
        for (int t = 0; t < TNUM; ++t) {
            float acc = base;
            for (int p = 0; p < PDIM; ++p) acc += mb2[(size_t)jj * PDIM + p] * emb[t * PDIM + p];
            bias2[t * DOUT + jj] = acc;
        }
    }
}

// ---------------------------------------------------------------------------
// 4) Big hypernetwork, HBM-streaming + WMMA:
//    A{1,2}[t] = bf16( W + mW_b + mW . e[t] ), written K-SWIZZLED:
//      plane index = ((kt * Ndim + j) * 32 + kk), kt=i/32, kk=i%32
//    so forward B-fragments are contiguous 32B per lane.
//    One wave: 16 flat rows x 8 tasks; K=64 -> 2x v_wmma_f32_16x16x32_bf16.
// ---------------------------------------------------------------------------
__global__ void mtms_hyper_kernel(const float* __restrict__ mW1, const float* __restrict__ mW1_b,
                                  const float* __restrict__ W1,
                                  const float* __restrict__ mW2, const float* __restrict__ mW2_b,
                                  const float* __restrict__ W2,
                                  const __bf16* __restrict__ eAfrag,
                                  __bf16* __restrict__ A1, __bf16* __restrict__ A2) {
    int wave = (blockIdx.x * blockDim.x + threadIdx.x) >> 5;
    int lane = threadIdx.x & 31;
    int which = wave >> 15;                          // 32768 row-tiles per matrix
    size_t rowBase = (size_t)(wave & 32767) * 16;
    const float* Mw = which ? mW2 : mW1;
    const float* hb = which ? mW2_b : mW1_b;
    const float* Wb = which ? W2 : W1;
    __bf16* Out = which ? A2 : A1;
    int nshift = which ? 9 : 10;                     // log2(Ndim)
    int ndim   = which ? DOUT : DH;

    int lane15 = lane & 15;
    int hiB = (lane >> 4) * 16;

    // Shared mesa A fragments (precomputed): one 32B load per K-tile
    const v16bf* eAv = (const v16bf*)eAfrag;
    v16bf a0 = eAv[lane];
    v16bf a1 = eAv[32 + lane];

    // B fragment: N=row (lane%16), K sequential (0..15 lo half-wave, 16..31 hi)
    const float* rp = Mw + (rowBase + lane15) * PDIM + hiB;
    v8f c = {};
    {
        const float4* q = (const float4*)rp;
        float4 q0 = q[0], q1 = q[1], q2 = q[2], q3 = q[3];
        v16bf b;
        b[0]  = (__bf16)q0.x; b[1]  = (__bf16)q0.y; b[2]  = (__bf16)q0.z; b[3]  = (__bf16)q0.w;
        b[4]  = (__bf16)q1.x; b[5]  = (__bf16)q1.y; b[6]  = (__bf16)q1.z; b[7]  = (__bf16)q1.w;
        b[8]  = (__bf16)q2.x; b[9]  = (__bf16)q2.y; b[10] = (__bf16)q2.z; b[11] = (__bf16)q2.w;
        b[12] = (__bf16)q3.x; b[13] = (__bf16)q3.y; b[14] = (__bf16)q3.z; b[15] = (__bf16)q3.w;
        c = __builtin_amdgcn_wmma_f32_16x16x32_bf16(false, a0, false, b, (short)0, c, false, false);
    }
    {
        const float4* q = (const float4*)(rp + 32);
        float4 q0 = q[0], q1 = q[1], q2 = q[2], q3 = q[3];
        v16bf b;
        b[0]  = (__bf16)q0.x; b[1]  = (__bf16)q0.y; b[2]  = (__bf16)q0.z; b[3]  = (__bf16)q0.w;
        b[4]  = (__bf16)q1.x; b[5]  = (__bf16)q1.y; b[6]  = (__bf16)q1.z; b[7]  = (__bf16)q1.w;
        b[8]  = (__bf16)q2.x; b[9]  = (__bf16)q2.y; b[10] = (__bf16)q2.z; b[11] = (__bf16)q2.w;
        b[12] = (__bf16)q3.x; b[13] = (__bf16)q3.y; b[14] = (__bf16)q3.z; b[15] = (__bf16)q3.w;
        c = __builtin_amdgcn_wmma_f32_16x16x32_bf16(false, a1, false, b, (short)0, c, false, false);
    }
    if (lane < 16) {                                  // lanes>=16 hold t=8..15 (invalid)
        size_t r = rowBase + lane15;                  // flat index into W / mW_b
        size_t i = rowBase >> nshift;                 // K index of this row-tile
        int j0 = (int)(rowBase & (size_t)(ndim - 1)); // N base of this row-tile
        int kt = (int)(i >> 5), kk = (int)(i & 31);
        float wv = Wb[r] + hb[r];
        size_t outIdx = ((size_t)kt * ndim + (size_t)(j0 + lane15)) * 32 + kk;
        #pragma unroll
        for (int k = 0; k < TNUM; ++k)                // D: M=t=k, N=row=lane
            Out[(size_t)k * PLANE + outIdx] = (__bf16)(c[k] + wv);
    }
}

// ---------------------------------------------------------------------------
// 5) Layer 1: H = relu(Xg @ A1[task] + bias1[task]); A1 is K-swizzled
// ---------------------------------------------------------------------------
__global__ void mtms_layer1_kernel(const __bf16* __restrict__ Xg,
                                   const __bf16* __restrict__ A1,
                                   const float* __restrict__ bias1,
                                   const int* __restrict__ tile_task,
                                   __bf16* __restrict__ H) {
    int wave = (blockIdx.x * blockDim.x + threadIdx.x) >> 5;
    int lane = threadIdx.x & 31;
    int rowTile = wave >> 6;          // DH/16 = 64 j-tiles
    int jTile = wave & 63;
    int task = tile_task[rowTile];
    if (task < 0) return;
    int rowBase = rowTile * 16, jBase = jTile * 16;
    int lane15 = lane & 15;
    int hiA = (lane >> 4) * 8;
    int hiB = (lane >> 4) * 16;
    const __bf16* Ap = Xg + (size_t)(rowBase + lane15) * DIN;
    const __bf16* Bp = A1 + (size_t)task * PLANE + (size_t)(jBase + lane15) * 32 + hiB;

    v8f c = {};
    #pragma unroll 4
    for (int kt = 0; kt < DIN / 32; ++kt) {
        v8bf lo = *(const v8bf*)(Ap + kt * 32 + hiB);
        v8bf hi = *(const v8bf*)(Ap + kt * 32 + hiB + 8);
        v16bf a = __builtin_shufflevector(lo, hi, 0,1,2,3,4,5,6,7,8,9,10,11,12,13,14,15);
        v16bf b = *(const v16bf*)(Bp + (size_t)kt * (DH * 32));
        c = __builtin_amdgcn_wmma_f32_16x16x32_bf16(false, a, false, b, (short)0, c, false, false);
    }
    float bias = bias1[task * DH + jBase + lane15];
    #pragma unroll
    for (int k = 0; k < 8; ++k) {
        float v = c[k] + bias;
        v = v > 0.f ? v : 0.f;
        H[(size_t)(rowBase + hiA + k) * DH + jBase + lane15] = (__bf16)v;
    }
}

// ---------------------------------------------------------------------------
// 6) Layer 2: Y = H @ A2[task] + bias2[task]; A2 K-swizzled; scatter via perm
// ---------------------------------------------------------------------------
__global__ void mtms_layer2_kernel(const __bf16* __restrict__ H,
                                   const __bf16* __restrict__ A2,
                                   const float* __restrict__ bias2,
                                   const int* __restrict__ tile_task,
                                   const int* __restrict__ perm,
                                   float* __restrict__ out) {
    int wave = (blockIdx.x * blockDim.x + threadIdx.x) >> 5;
    int lane = threadIdx.x & 31;
    int rowTile = wave >> 5;          // DOUT/16 = 32 j-tiles
    int jTile = wave & 31;
    int task = tile_task[rowTile];
    if (task < 0) return;
    int rowBase = rowTile * 16, jBase = jTile * 16;
    int lane15 = lane & 15;
    int hiA = (lane >> 4) * 8;
    int hiB = (lane >> 4) * 16;
    const __bf16* Ap = H + (size_t)(rowBase + lane15) * DH;
    const __bf16* Bp = A2 + (size_t)task * PLANE + (size_t)(jBase + lane15) * 32 + hiB;

    v8f c = {};
    #pragma unroll 4
    for (int kt = 0; kt < DH / 32; ++kt) {
        v8bf lo = *(const v8bf*)(Ap + kt * 32 + hiB);
        v8bf hi = *(const v8bf*)(Ap + kt * 32 + hiB + 8);
        v16bf a = __builtin_shufflevector(lo, hi, 0,1,2,3,4,5,6,7,8,9,10,11,12,13,14,15);
        v16bf b = *(const v16bf*)(Bp + (size_t)kt * (DOUT * 32));
        c = __builtin_amdgcn_wmma_f32_16x16x32_bf16(false, a, false, b, (short)0, c, false, false);
    }
    float bias = bias2[task * DOUT + jBase + lane15];
    #pragma unroll
    for (int k = 0; k < 8; ++k) {
        int pr = perm[rowBase + hiA + k];
        if (pr >= 0)
            out[(size_t)pr * DOUT + jBase + lane15] = c[k] + bias;
    }
}

// ---------------------------------------------------------------------------
// Launcher
// ---------------------------------------------------------------------------
extern "C" void kernel_launch(void* const* d_in, const int* in_sizes, int n_in,
                              void* d_out, int out_size, void* d_ws, size_t ws_size,
                              hipStream_t stream) {
    const float* x      = (const float*)d_in[0];
    const int*   tasks  = (const int*)  d_in[1];
    const float* W1     = (const float*)d_in[2];
    const float* b1     = (const float*)d_in[3];
    const float* W2     = (const float*)d_in[4];
    const float* b2     = (const float*)d_in[5];
    const float* mW1    = (const float*)d_in[6];
    const float* mW1_b  = (const float*)d_in[7];
    const float* mb1    = (const float*)d_in[8];
    const float* mb1_b  = (const float*)d_in[9];
    const float* mW2    = (const float*)d_in[10];
    const float* mW2_b  = (const float*)d_in[11];
    const float* mb2    = (const float*)d_in[12];
    const float* mb2_b  = (const float*)d_in[13];
    const float* emb    = (const float*)d_in[14];
    float* out = (float*)d_out;

    char* w = (char*)d_ws;
    __bf16* A1      = (__bf16*)(w);                  //  8 MiB: swizzled [T] planes
    __bf16* A2      = (__bf16*)(w + 8388608);        //  8 MiB: swizzled [T] planes
    __bf16* Xg      = (__bf16*)(w + 16777216);       //  1152*512*2
    __bf16* H       = (__bf16*)(w + 17956864);       //  1152*1024*2
    float*  bias1   = (float*) (w + 20316160);       //  T*DH*4
    float*  bias2   = (float*) (w + 20348928);       //  T*DOUT*4
    int*    perm    = (int*)   (w + 20365312);       //  1152*4
    int*    ttask   = (int*)   (w + 20369920);       //  NTILES*4
    __bf16* eAfrag  = (__bf16*)(w + 20370432);       //  64*16*2 = 2 KiB

    mtms_sort_kernel<<<1, 256, 0, stream>>>(tasks, perm, ttask);
    mtms_gather_kernel<<<(PAD_ROWS * DIN / 4) / 256, 256, 0, stream>>>(x, perm, Xg);
    mtms_bias_kernel<<<6, 256, 0, stream>>>(b1, b2, mb1, mb1_b, mb2, mb2_b, emb,
                                            bias1, bias2, eAfrag);
    // 2 matrices * 32768 row-tiles = 65536 waves, 8 waves/block
    mtms_hyper_kernel<<<8192, 256, 0, stream>>>(mW1, mW1_b, W1, mW2, mW2_b, W2,
                                                eAfrag, A1, A2);
    // 72 row-tiles * 64 j-tiles = 4608 waves
    mtms_layer1_kernel<<<576, 256, 0, stream>>>(Xg, A1, bias1, ttask, H);
    // 72 row-tiles * 32 j-tiles = 2304 waves
    mtms_layer2_kernel<<<288, 256, 0, stream>>>(H, A2, bias2, ttask, perm, out);
}
